// TinyRecursiveMoEAttnModel_80204219286165
// MI455X (gfx1250) — matrix operand order repaired and checked by reading, not compiled, not run on hardware
//
#include <hip/hip_runtime.h>
#include <hip/hip_bf16.h>
#include <math.h>
#include <stdint.h>

// ---------------- model dims ----------------
#define B_    4
#define T_    1024
#define V_    32000
#define D_    512
#define E_    8
#define HID_  1024
#define NH_   8
#define HD_   64
#define BT_   (B_*T_)      // 4096 tokens
#define STEPS_ 4

// ---------------- vector types ----------------
typedef __attribute__((ext_vector_type(16))) __bf16        v16bf;
typedef __attribute__((ext_vector_type(8)))  float         v8f;
typedef __attribute__((ext_vector_type(4)))  unsigned int  v4u;   // POD 16B chunk

union FragBF { v16bf v; v4u q[2]; };

static __device__ __forceinline__ __bf16 f2bf(float f) {
  union { float f; uint32_t u; } c; c.f = f;
  uint32_t r = (c.u + 0x7FFFu + ((c.u >> 16) & 1u)) >> 16;   // RNE
  unsigned short s = (unsigned short)r;
  return __builtin_bit_cast(__bf16, s);
}

// --------- CDNA5 async global->LDS copy (ASYNCcnt path, bypasses VGPR staging) ---------
// 32 bytes per lane: two b128 async loads; INST_OFFSET applies to both LDS and global
// addresses (CDNA5 ISA), so one base pair + offset:16 covers the chunk.
static __device__ __forceinline__ void async_copy32(unsigned lds_addr, const void* gaddr) {
  asm volatile(
      "global_load_async_to_lds_b128 %0, %1, off\n\t"
      "global_load_async_to_lds_b128 %0, %1, off offset:16"
      :: "v"(lds_addr), "v"(gaddr) : "memory");
}
static __device__ __forceinline__ void async_copy64(unsigned lds_addr, const void* gaddr) {
  asm volatile(
      "global_load_async_to_lds_b128 %0, %1, off\n\t"
      "global_load_async_to_lds_b128 %0, %1, off offset:16\n\t"
      "global_load_async_to_lds_b128 %0, %1, off offset:32\n\t"
      "global_load_async_to_lds_b128 %0, %1, off offset:48"
      :: "v"(lds_addr), "v"(gaddr) : "memory");
}
static __device__ __forceinline__ void wait_async0() {
  asm volatile("s_wait_asynccnt 0x0" ::: "memory");
}
// flat->LDS offset: aperture maps addr[31:0] to the wave's LDS allocation
static __device__ __forceinline__ unsigned lds_off(const void* p) {
  return (unsigned)(size_t)p;
}

// ---------------- f32 -> bf16 convert ----------------
__global__ void k_cvt_bf16(const float* __restrict__ in, __bf16* __restrict__ out, int n) {
  int i = blockIdx.x * 256 + threadIdx.x;
  if (i < n) out[i] = f2bf(in[i]);
}

// ---------------- embedding gather + state init ----------------
__global__ void k_embed(const int* __restrict__ ids, const float* __restrict__ embed_w,
                        float* __restrict__ x, float* __restrict__ latent,
                        float* __restrict__ pred) {
  int i = blockIdx.x * 256 + threadIdx.x;          // over BT_*D_
  int t = i >> 9, d = i & (D_ - 1);
  float v = embed_w[(size_t)ids[t] * D_ + d];
  x[i] = v; latent[i] = v; pred[i] = 0.f;
}

// ---------------- rmsnorm(latent+x)*w + prediction ; gate ----------------
__global__ __launch_bounds__(256)
void k_fuse_pre(const float* __restrict__ latent, const float* __restrict__ x,
                const float* __restrict__ pred, const float* __restrict__ rms_w,
                const float* __restrict__ sel_w, const float* __restrict__ sel_b,
                float* __restrict__ u, __bf16* __restrict__ u_bf, float* __restrict__ g) {
  const int t = blockIdx.x, tid = threadIdx.x;
  const size_t base = (size_t)t * D_;
  float y0 = latent[base + tid]       + x[base + tid];
  float y1 = latent[base + tid + 256] + x[base + tid + 256];
  float ss = y0*y0 + y1*y1;
  #pragma unroll
  for (int m = 16; m >= 1; m >>= 1) ss += __shfl_xor(ss, m, 32);
  __shared__ float red[8];
  if ((tid & 31) == 0) red[tid >> 5] = ss;
  __syncthreads();
  float sum = 0.f;
  #pragma unroll
  for (int i = 0; i < 8; ++i) sum += red[i];
  float inv = 1.f / (sqrtf(sum * (1.f / D_)) + 1e-8f);        // norm = ||y||/sqrt(D)
  float u0 = rms_w[tid]       * y0 * inv + pred[base + tid];
  float u1 = rms_w[tid + 256] * y1 * inv + pred[base + tid + 256];
  u[base + tid]       = u0;  u_bf[base + tid]       = f2bf(u0);
  u[base + tid + 256] = u1;  u_bf[base + tid + 256] = f2bf(u1);
  float dp = u0 * sel_w[tid] + u1 * sel_w[tid + 256];
  #pragma unroll
  for (int m = 16; m >= 1; m >>= 1) dp += __shfl_xor(dp, m, 32);
  __syncthreads();
  if ((tid & 31) == 0) red[tid >> 5] = dp;
  __syncthreads();
  if (tid == 0) {
    float tot = 0.f;
    for (int i = 0; i < 8; ++i) tot += red[i];
    g[t] = 1.f / (1.f + expf(-(tot + sel_b[0])));
  }
}

// ---------------- router logits + softmax (one token per block; wave e per expert) ----
__global__ __launch_bounds__(256)
void k_router(const float* __restrict__ u, const float* __restrict__ router_w,
              const float* __restrict__ router_b, float* __restrict__ rw) {
  const int t = blockIdx.x, w = threadIdx.x >> 5, l = threadIdx.x & 31;
  const float* ur = u + (size_t)t * D_;
  const float* wr = router_w + (size_t)w * D_;
  float s = 0.f;
  for (int d = l; d < D_; d += 32) s += ur[d] * wr[d];
  #pragma unroll
  for (int m = 16; m >= 1; m >>= 1) s += __shfl_xor(s, m, 32);
  __shared__ float lg[E_];
  if (l == 0) lg[w] = s + router_b[w];
  __syncthreads();
  if (threadIdx.x == 0) {
    float mx = lg[0];
    for (int e = 1; e < E_; ++e) mx = fmaxf(mx, lg[e]);
    float ex[E_], se = 0.f;
    for (int e = 0; e < E_; ++e) { ex[e] = expf(lg[e] - mx); se += ex[e]; }
    float inv = 1.f / se;
    for (int e = 0; e < E_; ++e) rw[(size_t)t * E_ + e] = ex[e] * inv;
  }
}

// ---------------- mix + latent update ----------------
__global__ void k_mix(const float* __restrict__ attn_out, const float* __restrict__ moe_out,
                      const float* __restrict__ g, const float* __restrict__ rw,
                      const float* __restrict__ e_b2, float* __restrict__ latent,
                      __bf16* __restrict__ latent_bf) {
  int i = blockIdx.x * 256 + threadIdx.x;
  int t = i >> 9, d = i & (D_ - 1);
  float gb = 0.f;                                    // sum_e w_e * b2[e][d]
  #pragma unroll
  for (int e = 0; e < E_; ++e) gb += rw[(size_t)t * E_ + e] * e_b2[(size_t)e * D_ + d];
  float gg = g[t];
  float mix = gg * attn_out[i] + (1.f - gg) * (moe_out[i] + gb);
  float nl = latent[i] + mix;
  latent[i] = nl;
  latent_bf[i] = f2bf(nl);
}

// ================= generic bf16 WMMA GEMM:  C[M,N] = A[M,K] * W[N,K]^T =================
#define BM 128
#define BN 128
#define BK 32
#define LDT 40   // LDS row stride (halfwords): 32 + 8 pad, rows stay 16B aligned

constexpr int EP_F32            = 0;  // store f32
constexpr int EP_BF16           = 1;  // store bf16
constexpr int EP_BIAS_F32_BF16  = 2;  // + bias[n], store f32 and bf16
constexpr int EP_GELU_SCALE_BF16= 3;  // gelu(acc+bias)*scale[row,E=z] -> bf16

template <int EP, bool MOE2>
__global__ __launch_bounds__(256)
void k_gemm(const __bf16* __restrict__ A, const __bf16* __restrict__ Bw,
            float* __restrict__ Cf, __bf16* __restrict__ Cb,
            const float* __restrict__ bias, const float* __restrict__ scale,
            int M, int N, int K, int KperE,
            long long bStrideZ, long long biasStrideZ, long long cbStrideZ) {
  __shared__ __bf16 As[2][BM * LDT];
  __shared__ __bf16 Bs[2][BN * LDT];

  const int tid  = threadIdx.x;
  const int lane = tid & 31, wv = tid >> 5;
  const int wM = wv >> 1, wN = wv & 1;            // 4 (M) x 2 (N) waves
  const int hf = lane >> 4, l16 = lane & 15;
  const int z = blockIdx.z;
  const int tileM = blockIdx.y * BM;
  const int tileN = blockIdx.x * BN;

  const __bf16* Bz = Bw + (size_t)z * (size_t)bStrideZ;

  const int lrow = tid >> 1;                      // 0..127
  const int lseg = tid & 1;                       // 16 halfwords (32B) each
  const unsigned ldsA0 = lds_off(&As[0][lrow * LDT + lseg * 16]);
  const unsigned ldsA1 = lds_off(&As[1][lrow * LDT + lseg * 16]);
  const unsigned ldsB0 = lds_off(&Bs[0][lrow * LDT + lseg * 16]);
  const unsigned ldsB1 = lds_off(&Bs[1][lrow * LDT + lseg * 16]);

  v8f acc[2][4];
  v8f vzero = {};
  #pragma unroll
  for (int i = 0; i < 2; ++i)
    #pragma unroll
    for (int j = 0; j < 4; ++j) acc[i][j] = vzero;

  const int nk = K / BK;

  auto asyncTile = [&](int kt, int bufSel) {
    int k0 = kt * BK;
    const __bf16 *arow, *brow;
    if constexpr (MOE2) {        // A: [e][M][KperE], W: [e][N][KperE], sum over experts
      int e = k0 / KperE, kin = k0 % KperE;
      arow = A  + ((size_t)e * M + (size_t)(tileM + lrow)) * KperE + kin + lseg * 16;
      brow = Bw + ((size_t)e * N + (size_t)(tileN + lrow)) * KperE + kin + lseg * 16;
    } else {
      arow = A  + (size_t)(tileM + lrow) * K + k0 + lseg * 16;
      brow = Bz + (size_t)(tileN + lrow) * K + k0 + lseg * 16;
    }
    async_copy32(bufSel ? ldsA1 : ldsA0, arow);
    async_copy32(bufSel ? ldsB1 : ldsB0, brow);
  };

  int buf = 0;
  asyncTile(0, 0);
  wait_async0();
  __syncthreads();

  for (int kt = 0; kt < nk; ++kt) {
    if (kt + 1 < nk) asyncTile(kt + 1, buf ^ 1);   // in flight during compute

    FragBF af[2], bfr[4];
    // A frag (16-bit A 16x32): lane row = l16; elems 0..15 = k[hf*16..+15]
    #pragma unroll
    for (int fm = 0; fm < 2; ++fm) {
      const __bf16* p = &As[buf][(wM * 32 + fm * 16 + l16) * LDT];
      af[fm].q[0] = *(const v4u*)(p + hf * 16);
      af[fm].q[1] = *(const v4u*)(p + hf * 16 + 8);
    }
    // B frag: lane holds W[n = l16 of tile][k = hf*16 .. +15] (B[k][n] = W[n][k])
    #pragma unroll
    for (int fn = 0; fn < 4; ++fn) {
      const __bf16* p = &Bs[buf][(wN * 64 + fn * 16 + l16) * LDT + hf * 16];
      bfr[fn].q[0] = ((const v4u*)p)[0];
      bfr[fn].q[1] = ((const v4u*)p)[1];
    }
    #pragma unroll
    for (int fm = 0; fm < 2; ++fm)
      #pragma unroll
      for (int fn = 0; fn < 4; ++fn)
        acc[fm][fn] = __builtin_amdgcn_wmma_f32_16x16x32_bf16(
            false, af[fm].v, false, bfr[fn].v, (short)0, acc[fm][fn], false, false);

    wait_async0();       // our next-tile async copies have landed
    __syncthreads();     // everyone's copies visible; everyone done reading buf
    buf ^= 1;
  }

  // epilogue: C row = hf*8 + r, col = l16 within each 16x16 frag
  const float* biasZ = bias ? bias + (size_t)z * (size_t)biasStrideZ : nullptr;
  __bf16* CbZ = Cb ? Cb + (size_t)z * (size_t)cbStrideZ : nullptr;
  #pragma unroll
  for (int fm = 0; fm < 2; ++fm) {
    #pragma unroll
    for (int fn = 0; fn < 4; ++fn) {
      int gn = tileN + wN * 64 + fn * 16 + l16;
      #pragma unroll
      for (int r = 0; r < 8; ++r) {
        int gm = tileM + wM * 32 + fm * 16 + hf * 8 + r;
        float v = acc[fm][fn][r];
        if constexpr (EP == EP_F32) {
          Cf[(size_t)gm * N + gn] = v;
        } else if constexpr (EP == EP_BF16) {
          CbZ[(size_t)gm * N + gn] = f2bf(v);
        } else if constexpr (EP == EP_BIAS_F32_BF16) {
          v += biasZ[gn];
          Cf[(size_t)gm * N + gn] = v;
          CbZ[(size_t)gm * N + gn] = f2bf(v);
        } else {  // EP_GELU_SCALE_BF16
          v += biasZ[gn];
          v = 0.5f * v * (1.f + erff(v * 0.70710678118654752f));   // exact GELU
          v *= scale[(size_t)gm * E_ + z];                          // router weight
          CbZ[(size_t)gm * N + gn] = f2bf(v);
        }
      }
    }
  }
}

// ================= flash attention (bf16 WMMA, online softmax) =================
// grid: (T/128, NH, B); block 256 = 8 waves; wave w owns q rows [w*16, w*16+16)
__global__ __launch_bounds__(256)
void k_attn(const __bf16* __restrict__ qkv, __bf16* __restrict__ obf) {
  __shared__ __bf16 Qs[128 * 72];
  __shared__ __bf16 Ks[64 * 72];
  __shared__ __bf16 Vts[64 * 72];          // V transposed: [hd][key]
  __shared__ __bf16 Ps[8][16 * 72];        // per-wave P scratch

  const int tid = threadIdx.x, lane = tid & 31, wv = tid >> 5;
  const int hf = lane >> 4, l16 = lane & 15;
  const int qbase = blockIdx.x * 128;
  const int h = blockIdx.y, b = blockIdx.z;
  const size_t rs = 3 * D_;                // qkv row stride (halfwords)
  const __bf16* qp = qkv + (size_t)(b * T_) * rs + h * HD_;
  const __bf16* kp = qp + D_;
  const __bf16* vp = qp + 2 * D_;

  // Q tile: 128 x 64 via async copy (64B per lane)
  {
    int row = tid >> 1, seg = tid & 1;
    async_copy64(lds_off(&Qs[row * 72 + seg * 32]),
                 qp + (size_t)(qbase + row) * rs + seg * 32);
  }

  const int krow = tid >> 2, kseg = tid & 3;
  const unsigned ldsK = lds_off(&Ks[krow * 72 + kseg * 16]);

  float mrow[8], lsum[8];
  v8f ao[4]; v8f vzero = {};
  #pragma unroll
  for (int r = 0; r < 8; ++r) { mrow[r] = -INFINITY; lsum[r] = 0.f; }
  #pragma unroll
  for (int j = 0; j < 4; ++j) ao[j] = vzero;

  for (int kt = 0; kt < T_ / 64; ++kt) {
    __syncthreads();                        // done reading Ks/Vts of previous tile
    // K tile: async; V tile: manual transpose into Vts
    async_copy32(ldsK, kp + (size_t)(kt * 64 + krow) * rs + kseg * 16);
    {
      const __bf16* vs = vp + (size_t)(kt * 64 + krow) * rs + kseg * 16;
      FragBF vvf;
      vvf.q[0] = ((const v4u*)vs)[0];
      vvf.q[1] = ((const v4u*)vs)[1];
      #pragma unroll
      for (int j = 0; j < 16; ++j) Vts[(kseg * 16 + j) * 72 + krow] = vvf.v[j];
    }
    wait_async0();                          // Q (first iter) + K landed
    __syncthreads();

    // S = Q * K^T  (M=16 q rows, N=64 keys, contraction = HD)
    v8f sfr[4];
    #pragma unroll
    for (int j = 0; j < 4; ++j) sfr[j] = vzero;
    #pragma unroll
    for (int kk = 0; kk < 64; kk += 32) {
      FragBF aq;
      const __bf16* p = &Qs[(wv * 16 + l16) * 72 + kk];
      aq.q[0] = *(const v4u*)(p + hf * 16);
      aq.q[1] = *(const v4u*)(p + hf * 16 + 8);
      #pragma unroll
      for (int fn = 0; fn < 4; ++fn) {
        FragBF bk;
        const __bf16* pb = &Ks[(fn * 16 + l16) * 72 + kk + hf * 16];
        bk.q[0] = ((const v4u*)pb)[0];
        bk.q[1] = ((const v4u*)pb)[1];
        sfr[fn] = __builtin_amdgcn_wmma_f32_16x16x32_bf16(
            false, aq.v, false, bk.v, (short)0, sfr[fn], false, false);
      }
    }
    #pragma unroll
    for (int fn = 0; fn < 4; ++fn)
      #pragma unroll
      for (int r = 0; r < 8; ++r) sfr[fn][r] *= 0.125f;     // 1/sqrt(HD)

    // online softmax per row (row = hf*8 + r; reduce over 16 lanes of this half)
    #pragma unroll
    for (int r = 0; r < 8; ++r) {
      float mx = fmaxf(fmaxf(sfr[0][r], sfr[1][r]), fmaxf(sfr[2][r], sfr[3][r]));
      #pragma unroll
      for (int m = 8; m >= 1; m >>= 1) mx = fmaxf(mx, __shfl_xor(mx, m, 32));
      float mnew = fmaxf(mrow[r], mx);
      float alpha = expf(mrow[r] - mnew);
      mrow[r] = mnew;
      float ps = 0.f;
      #pragma unroll
      for (int fn = 0; fn < 4; ++fn) {
        float p = expf(sfr[fn][r] - mnew);
        sfr[fn][r] = p; ps += p;
      }
      #pragma unroll
      for (int m = 8; m >= 1; m >>= 1) ps += __shfl_xor(ps, m, 32);
      lsum[r] = lsum[r] * alpha + ps;
      #pragma unroll
      for (int fn = 0; fn < 4; ++fn) ao[fn][r] *= alpha;
    }

    // P -> per-wave LDS (re-layout as A fragment), then O += P * V
    #pragma unroll
    for (int fn = 0; fn < 4; ++fn)
      #pragma unroll
      for (int r = 0; r < 8; ++r)
        Ps[wv][(hf * 8 + r) * 72 + fn * 16 + l16] = f2bf(sfr[fn][r]);

    #pragma unroll
    for (int kk = 0; kk < 64; kk += 32) {
      FragBF ap;
      const __bf16* p = &Ps[wv][l16 * 72 + kk];
      ap.q[0] = *(const v4u*)(p + hf * 16);
      ap.q[1] = *(const v4u*)(p + hf * 16 + 8);
      #pragma unroll
      for (int fn = 0; fn < 4; ++fn) {
        FragBF bv;
        const __bf16* pb = &Vts[(fn * 16 + l16) * 72 + kk + hf * 16];
        bv.q[0] = ((const v4u*)pb)[0];
        bv.q[1] = ((const v4u*)pb)[1];
        ao[fn] = __builtin_amdgcn_wmma_f32_16x16x32_bf16(
            false, ap.v, false, bv.v, (short)0, ao[fn], false, false);
      }
    }
  }

  // epilogue: o[tok][h*HD + n] = O / l
  #pragma unroll
  for (int r = 0; r < 8; ++r) {
    float inv = 1.f / lsum[r];
    int tok = b * T_ + qbase + wv * 16 + hf * 8 + r;
    __bf16* orow = obf + (size_t)tok * D_ + h * HD_;
    #pragma unroll
    for (int fn = 0; fn < 4; ++fn)
      orow[fn * 16 + l16] = f2bf(ao[fn][r] * inv);
  }
}

// ================= host orchestration =================
extern "C" void kernel_launch(void* const* d_in, const int* in_sizes, int n_in,
                              void* d_out, int out_size, void* d_ws, size_t ws_size,
                              hipStream_t stream) {
  const int*   ids      = (const int*)  d_in[0];
  const float* embed_w  = (const float*)d_in[1];
  const float* rms_w    = (const float*)d_in[2];
  const float* sel_w    = (const float*)d_in[3];
  const float* sel_b    = (const float*)d_in[4];
  const float* qkv_w    = (const float*)d_in[5];
  const float* out_w    = (const float*)d_in[6];
  const float* router_w = (const float*)d_in[7];
  const float* router_b = (const float*)d_in[8];
  const float* e_w1     = (const float*)d_in[9];
  const float* e_b1     = (const float*)d_in[10];
  const float* e_w2     = (const float*)d_in[11];
  const float* e_b2     = (const float*)d_in[12];
  const float* pred_w   = (const float*)d_in[13];
  const float* pred_b   = (const float*)d_in[14];
  const float* rev_w    = (const float*)d_in[15];
  float* out = (float*)d_out;

  char* ws = (char*)d_ws;
  size_t off = 0;
  auto alloc = [&](size_t bytes) -> void* {
    void* p = (void*)(ws + off);
    off += (bytes + 255) & ~(size_t)255;
    return p;
  };

  __bf16* qkv_w_bf  = (__bf16*)alloc((size_t)3 * D_ * D_ * 2);
  __bf16* out_w_bf  = (__bf16*)alloc((size_t)D_ * D_ * 2);
  __bf16* e_w1_bf   = (__bf16*)alloc((size_t)E_ * HID_ * D_ * 2);
  __bf16* e_w2_bf   = (__bf16*)alloc((size_t)E_ * D_ * HID_ * 2);
  __bf16* pred_w_bf = (__bf16*)alloc((size_t)D_ * D_ * 2);
  __bf16* rev_w_bf  = (__bf16*)alloc((size_t)V_ * D_ * 2);
  float*  x         = (float*) alloc((size_t)BT_ * D_ * 4);
  float*  latent    = (float*) alloc((size_t)BT_ * D_ * 4);
  __bf16* latent_bf = (__bf16*)alloc((size_t)BT_ * D_ * 2);
  float*  pred      = (float*) alloc((size_t)BT_ * D_ * 4);
  __bf16* pred_bf   = (__bf16*)alloc((size_t)BT_ * D_ * 2);
  float*  u         = (float*) alloc((size_t)BT_ * D_ * 4);
  __bf16* u_bf      = (__bf16*)alloc((size_t)BT_ * D_ * 2);
  float*  g         = (float*) alloc((size_t)BT_ * 4);
  float*  rw        = (float*) alloc((size_t)BT_ * E_ * 4);
  __bf16* qkvb      = (__bf16*)alloc((size_t)BT_ * 3 * D_ * 2);
  __bf16* obf       = (__bf16*)alloc((size_t)BT_ * D_ * 2);
  float*  attn_out  = (float*) alloc((size_t)BT_ * D_ * 4);
  __bf16* hbf       = (__bf16*)alloc((size_t)E_ * BT_ * HID_ * 2);
  float*  moe_out   = (float*) alloc((size_t)BT_ * D_ * 4);
  (void)ws_size; (void)in_sizes; (void)n_in; (void)out_size;

  auto cvt = [&](const float* s, __bf16* d, size_t n) {
    k_cvt_bf16<<<(unsigned)((n + 255) / 256), 256, 0, stream>>>(s, d, (int)n);
  };
  cvt(qkv_w,  qkv_w_bf,  (size_t)3 * D_ * D_);
  cvt(out_w,  out_w_bf,  (size_t)D_ * D_);
  cvt(e_w1,   e_w1_bf,   (size_t)E_ * HID_ * D_);
  cvt(e_w2,   e_w2_bf,   (size_t)E_ * D_ * HID_);
  cvt(pred_w, pred_w_bf, (size_t)D_ * D_);
  cvt(rev_w,  rev_w_bf,  (size_t)V_ * D_);

  k_embed<<<BT_ * D_ / 256, 256, 0, stream>>>(ids, embed_w, x, latent, pred);

  for (int s = 0; s < STEPS_; ++s) {
    k_fuse_pre<<<BT_, 256, 0, stream>>>(latent, x, pred, rms_w, sel_w, sel_b, u, u_bf, g);
    k_router<<<BT_, 256, 0, stream>>>(u, router_w, router_b, rw);

    // qkv = u @ qkv_w^T  -> bf16
    k_gemm<EP_BF16, false><<<dim3(3 * D_ / BN, BT_ / BM, 1), 256, 0, stream>>>(
        u_bf, qkv_w_bf, nullptr, qkvb, nullptr, nullptr, BT_, 3 * D_, D_, 1, 0, 0, 0);

    k_attn<<<dim3(T_ / 128, NH_, B_), 256, 0, stream>>>(qkvb, obf);

    // attn_out = o @ out_w^T
    k_gemm<EP_F32, false><<<dim3(D_ / BN, BT_ / BM, 1), 256, 0, stream>>>(
        obf, out_w_bf, attn_out, nullptr, nullptr, nullptr, BT_, D_, D_, 1, 0, 0, 0);

    // h[e] = rw[:,e] * gelu(u @ w1[e]^T + b1[e])   (batched over grid.z = expert)
    k_gemm<EP_GELU_SCALE_BF16, false><<<dim3(HID_ / BN, BT_ / BM, E_), 256, 0, stream>>>(
        u_bf, e_w1_bf, nullptr, hbf, e_b1, rw, BT_, HID_, D_, 1,
        (long long)HID_ * D_, (long long)HID_, (long long)BT_ * HID_);

    // moe_out = sum_e h[e] @ w2[e]^T   (expert-summed K loop)
    k_gemm<EP_F32, true><<<dim3(D_ / BN, BT_ / BM, 1), 256, 0, stream>>>(
        hbf, e_w2_bf, moe_out, nullptr, nullptr, nullptr, BT_, D_, E_ * HID_, HID_, 0, 0, 0);

    k_mix<<<BT_ * D_ / 256, 256, 0, stream>>>(attn_out, moe_out, g, rw, e_b2, latent, latent_bf);

    // prediction = latent @ pred_w^T + pred_b  (f32 + bf16 copies)
    k_gemm<EP_BIAS_F32_BF16, false><<<dim3(D_ / BN, BT_ / BM, 1), 256, 0, stream>>>(
        latent_bf, pred_w_bf, pred, pred_bf, pred_b, nullptr, BT_, D_, D_, 1, 0, 1, 0);
  }

  // logits = prediction @ rev_w^T   (M=4096, N=32000, K=512)
  k_gemm<EP_F32, false><<<dim3(V_ / BN, BT_ / BM, 1), 256, 0, stream>>>(
      pred_bf, rev_w_bf, out, nullptr, nullptr, nullptr, BT_, V_, D_, 1, 0, 0, 0);
}
